// SievedLinear_79035988181284
// MI455X (gfx1250) — compile-verified, matrix operand order checked by reading, not run on hardware
//
#include <hip/hip_runtime.h>
#include <hip/hip_bf16.h>

typedef __bf16 bf16_t;
typedef __attribute__((ext_vector_type(16))) __bf16 v16bf;
typedef __attribute__((ext_vector_type(8)))  __bf16 v8bf;
typedef __attribute__((ext_vector_type(8)))  float  v8f;
typedef __attribute__((__vector_size__(16))) int vi4;   // matches builtin param pointee

#ifndef __has_builtin
#define __has_builtin(x) 0
#endif

// ---- CDNA5 async global->LDS path (guarded so host pass / older clang still compile) ----
#if defined(__HIP_DEVICE_COMPILE__) && __has_builtin(__builtin_amdgcn_global_load_async_to_lds_b128)
#define ASYNC_COPY 1
#else
#define ASYNC_COPY 0
#endif

#if ASYNC_COPY
#if __has_builtin(__builtin_amdgcn_s_wait_asynccnt)
#define WAIT_ASYNC(n) __builtin_amdgcn_s_wait_asynccnt(n)
#else
#define WAIT_ASYNC(n) asm volatile("s_wait_asynccnt %0" ::"n"(n) : "memory")
#endif
#else
#define WAIT_ASYNC(n) do {} while (0)
#endif

__device__ __forceinline__ void copy16(const bf16_t* g, bf16_t* l) {
#if ASYNC_COPY
  __builtin_amdgcn_global_load_async_to_lds_b128(
      (__attribute__((address_space(1))) vi4*)(void*)g,
      (__attribute__((address_space(3))) vi4*)(void*)l,
      /*offset=*/0, /*cpol=*/0);
#else
  *(v8bf*)l = *(const v8bf*)g;   // synchronous fallback (still b128)
#endif
}

// =====================================================================
// Kernel 1: FWHT over a 4096-float row in LDS, scale by 1/sqrt(4096),
// emit bf16. One workgroup (256 threads, 8 waves) per row.
// =====================================================================
__global__ __launch_bounds__(256) void fwht4096_to_bf16(
    const float* __restrict__ x, bf16_t* __restrict__ xt) {
  __shared__ __attribute__((aligned(16))) float s[4096];
  const int tid = threadIdx.x;
  const float4* r4 = (const float4*)(x + (size_t)blockIdx.x * 4096);
  float4* s4 = (float4*)s;
#pragma unroll
  for (int i = tid; i < 1024; i += 256) s4[i] = r4[i];
  __syncthreads();

  for (int h = 1; h < 4096; h <<= 1) {
#pragma unroll
    for (int p = tid; p < 2048; p += 256) {
      int i = ((p & ~(h - 1)) << 1) | (p & (h - 1));
      float a = s[i], b = s[i + h];
      s[i] = a + b;
      s[i + h] = a - b;
    }
    __syncthreads();
  }

  bf16_t* orow = xt + (size_t)blockIdx.x * 4096;
#pragma unroll
  for (int i = tid; i < 512; i += 256) {   // 512 groups of 8 bf16 (b128 stores)
    v8bf o;
#pragma unroll
    for (int j = 0; j < 8; ++j) o[j] = (bf16_t)(s[i * 8 + j] * 0.015625f);
    *(v8bf*)(orow + i * 8) = o;
  }
}

// =====================================================================
// Kernel 2: weight fp32 -> bf16 (memory-bound, trivial)
// =====================================================================
__global__ __launch_bounds__(256) void cvt_f32_to_bf16(
    const float* __restrict__ w, bf16_t* __restrict__ wb, size_t n8) {
  size_t i = (size_t)blockIdx.x * blockDim.x + threadIdx.x;
  if (i >= n8) return;
  size_t base = i * 8;
  float4 f0 = *(const float4*)(w + base);
  float4 f1 = *(const float4*)(w + base + 4);
  v8bf o;
  o[0] = (bf16_t)f0.x; o[1] = (bf16_t)f0.y; o[2] = (bf16_t)f0.z; o[3] = (bf16_t)f0.w;
  o[4] = (bf16_t)f1.x; o[5] = (bf16_t)f1.y; o[6] = (bf16_t)f1.z; o[7] = (bf16_t)f1.w;
  *(v8bf*)(wb + base) = o;
}

// =====================================================================
// Kernel 3: bf16 WMMA GEMM, C[m,n] = sum_k A[m,k] * W[n,k]
//   BM=BN=128, BK=32; 256 threads = 8 waves in a 4(m) x 2(n) grid;
//   each wave: 2x4 tiles of v_wmma_f32_16x16x32_bf16.
//   LDS double-buffered, row stride 40 bf16 (80 B: 16B-aligned,
//   conflict-free b128 fragment reads).
// =====================================================================
#define BM 128
#define BN 128
#define BK 32
#define LDSS 40          // bf16 elements per LDS row (80 bytes)
#define KTILE_LOADS 4    // async b128 issues per thread per k-tile (2 A + 2 B)

__device__ __forceinline__ v16bf load_frag(const bf16_t* base) {
  // base points at (row, khalf); chunks at +0 and +16 elements per ISA layout
  v8bf lo = *(const v8bf*)(base);
  v8bf hi = *(const v8bf*)(base + 16);
  v16bf r;
#pragma unroll
  for (int i = 0; i < 8; ++i) { r[i] = lo[i]; r[i + 8] = hi[i]; }
  return r;
}

__global__ __launch_bounds__(256, 2) void gemm_bf16_wmma(
    const bf16_t* __restrict__ A,   // [M, K] row-major (FWHT output)
    const bf16_t* __restrict__ W,   // [N, K] row-major (weight)
    float* __restrict__ C,          // [M, N] row-major
    int M, int N, int K) {
  __shared__ __attribute__((aligned(16))) bf16_t sA[2][BM * LDSS];
  __shared__ __attribute__((aligned(16))) bf16_t sB[2][BN * LDSS];

  const int tid  = threadIdx.x;
  const int lane = tid & 31;
  const int wave = tid >> 5;
  const int wm = wave & 3;              // 0..3 -> 32-row slab
  const int wn = wave >> 2;             // 0..1 -> 64-col slab
  const int lrow  = lane & 15;
  const int khalf = (lane >> 4) << 3;   // 0 or 8 (ISA 16-bit A/B lane layout)

  const int m_base = blockIdx.y * BM;
  const int n_base = blockIdx.x * BN;
  const bf16_t* gA = A + (size_t)m_base * K;
  const bf16_t* gW = W + (size_t)n_base * K;

  v8f acc[2][4];
#pragma unroll
  for (int i = 0; i < 2; ++i)
#pragma unroll
    for (int j = 0; j < 4; ++j)
#pragma unroll
      for (int e = 0; e < 8; ++e) acc[i][j][e] = 0.0f;

  const int KT = K / BK;   // 128

  // ---- tile copy: 128 rows x 32 bf16 = 512 b128 transfers; 2 per thread ----
  auto copy_tile = [&](int kt, int buf) {
    const bf16_t* ga = gA + kt * BK;
    const bf16_t* gw = gW + kt * BK;
#pragma unroll
    for (int i = tid; i < 512; i += 256) {
      int row = i >> 2, chunk = i & 3;
      copy16(ga + (size_t)row * K + chunk * 8, &sA[buf][row * LDSS + chunk * 8]);
    }
#pragma unroll
    for (int i = tid; i < 512; i += 256) {
      int row = i >> 2, chunk = i & 3;
      copy16(gw + (size_t)row * K + chunk * 8, &sB[buf][row * LDSS + chunk * 8]);
    }
  };

  copy_tile(0, 0);                       // prologue

  for (int kt = 0; kt < KT; ++kt) {
    const int cur = kt & 1;
    if (kt + 1 < KT) {
      copy_tile(kt + 1, cur ^ 1);        // prefetch next tile into other buffer
      WAIT_ASYNC(KTILE_LOADS);           // tile kt landed; kt+1 may stay in flight
    } else {
      WAIT_ASYNC(0);
    }
    __syncthreads();

    v16bf af[2], bf[4];
#pragma unroll
    for (int mt = 0; mt < 2; ++mt)
      af[mt] = load_frag(&sA[cur][(wm * 32 + mt * 16 + lrow) * LDSS + khalf]);
#pragma unroll
    for (int nt = 0; nt < 4; ++nt)
      bf[nt] = load_frag(&sB[cur][(wn * 64 + nt * 16 + lrow) * LDSS + khalf]);

#pragma unroll
    for (int mt = 0; mt < 2; ++mt)
#pragma unroll
      for (int nt = 0; nt < 4; ++nt)
        acc[mt][nt] = __builtin_amdgcn_wmma_f32_16x16x32_bf16(
            false, af[mt], false, bf[nt], (short)0, acc[mt][nt], false, false);

    __syncthreads();                     // protect buffer being refilled next iter
  }

  // ---- epilogue: C/D layout = VGPR v -> (M = v + 8*(lane>=16), N = lane&15) ----
  const int mhi = (lane >> 4) << 3;
#pragma unroll
  for (int mt = 0; mt < 2; ++mt)
#pragma unroll
    for (int nt = 0; nt < 4; ++nt) {
      int n = n_base + wn * 64 + nt * 16 + (lane & 15);
#pragma unroll
      for (int v = 0; v < 8; ++v) {
        int m = m_base + wm * 32 + mt * 16 + mhi + v;
        C[(size_t)m * N + n] = acc[mt][nt][v];
      }
    }
}

// =====================================================================
// Launch: FWHT -> weight convert -> WMMA GEMM (all on `stream`).
// d_ws layout: [ M*D bf16 : 128 MB ][ N*D bf16 : 32 MB ]
// =====================================================================
extern "C" void kernel_launch(void* const* d_in, const int* in_sizes, int n_in,
                              void* d_out, int out_size, void* d_ws, size_t ws_size,
                              hipStream_t stream) {
  const float* x = (const float*)d_in[0];
  const float* w = (const float*)d_in[1];
  float* out = (float*)d_out;

  const int D = 4096;                    // D_in == D_out == 4096 per reference
  const int M = in_sizes[0] / D;         // B*S = 16384
  const int N = in_sizes[1] / D;         // 4096

  bf16_t* xt = (bf16_t*)d_ws;
  bf16_t* wb = (bf16_t*)((char*)d_ws + (size_t)M * D * sizeof(bf16_t));

  fwht4096_to_bf16<<<M, 256, 0, stream>>>(x, xt);

  size_t n8 = (size_t)N * D / 8;
  cvt_f32_to_bf16<<<(int)((n8 + 255) / 256), 256, 0, stream>>>(w, wb, n8);

  dim3 grid(N / BN, M / BM);             // 32 x 128
  gemm_bf16_wmma<<<grid, 256, 0, stream>>>(xt, wb, out, M, N, D);
}
